// MultivariateGaussian_51350628991245
// MI455X (gfx1250) — matrix operand order repaired and checked by reading, not compiled, not run on hardware
//
#include <hip/hip_runtime.h>
#include <hip/hip_bf16.h>
#include <cmath>

#define NN 8192
#define DD 512
#define BK 32
#define LDA 40   // padded LDS row stride in bf16 elems (80 B, keeps 16B alignment, spreads banks)
#define NXD (NN * DD)

typedef __attribute__((ext_vector_type(16))) __bf16 v16bf;
typedef __attribute__((ext_vector_type(8)))  __bf16 v8bf;
typedef __attribute__((ext_vector_type(4)))  __bf16 v4bf;
typedef __attribute__((ext_vector_type(8)))  float  v8f;

__device__ __forceinline__ void split2(float v, __bf16& h, __bf16& l) {
    h = (__bf16)v;
    l = (__bf16)(v - (float)h);
}

// A fragment (16-bit A 16x32): lanes 0-15 hold K 0..7 & 16..23; lanes 16-31 hold K 8..15 & 24..31
__device__ __forceinline__ v16bf load_fragA(const __bf16* rowp, int lane) {
    const int s = (lane >= 16) ? 8 : 0;
    v8bf lo = *(const v8bf*)(rowp + s);
    v8bf hi = *(const v8bf*)(rowp + s + 16);
    v16bf r;
#pragma unroll
    for (int i = 0; i < 8; ++i) { r[i] = lo[i]; r[i + 8] = hi[i]; }
    return r;
}

// B fragment (16-bit B 32x16): lanes 0-15 hold K 0..7 & 16..23; lanes 16-31 hold K 8..15 & 24..31
__device__ __forceinline__ v16bf load_fragB(const __bf16* rowp, int lane) {
    const int s = (lane >= 16) ? 8 : 0;
    v8bf lo = *(const v8bf*)(rowp + s);
    v8bf hi = *(const v8bf*)(rowp + s + 16);
    v16bf r;
#pragma unroll
    for (int i = 0; i < 8; ++i) { r[i] = lo[i]; r[i + 8] = hi[i]; }
    return r;
}

// ---------- one-time fp32 -> split-bf16 conversion: xh,xl and (x*rho)h,(x*rho)l ----------
__global__ __launch_bounds__(256) void prep_kernel(
    const float* __restrict__ x,
    const float* __restrict__ rho,
    __bf16* __restrict__ xh, __bf16* __restrict__ xl,
    __bf16* __restrict__ xrh, __bf16* __restrict__ xrl) {
    const size_t t = (size_t)blockIdx.x * blockDim.x + threadIdx.x;
    const size_t base = t * 4;                 // 4 consecutive elements, same row
    const int k = (int)(base & (DD - 1));      // column within row (multiple of 4)
    const float4 xv = *(const float4*)(x + base);
    const float4 rv = *(const float4*)(rho + k);
    float xs[4] = {xv.x, xv.y, xv.z, xv.w};
    float rs[4] = {xv.x * rv.x, xv.y * rv.y, xv.z * rv.z, xv.w * rv.w};
    v4bf h, l, rh, rl;
#pragma unroll
    for (int i = 0; i < 4; ++i) {
        __bf16 a, b;
        split2(xs[i], a, b); h[i] = a; l[i] = b;
        split2(rs[i], a, b); rh[i] = a; rl[i] = b;
    }
    *(v4bf*)(xh + base)  = h;
    *(v4bf*)(xl + base)  = l;
    *(v4bf*)(xrh + base) = rh;
    *(v4bf*)(xrl + base) = rl;
}

// ---------------- mu / diag-variance head (memory-bound GEMVs) ----------------
__global__ __launch_bounds__(256) void head_kernel(
    const float* __restrict__ x,
    const float* __restrict__ muk,
    const float* __restrict__ vark,
    const float* __restrict__ mu_bias,
    const float* __restrict__ var_bias,
    float* __restrict__ mu_out,
    float* __restrict__ d_ws) {
    const int wave = threadIdx.x >> 5;
    const int lane = threadIdx.x & 31;
    const int row = blockIdx.x * 8 + wave;
    const float* xr = x + (size_t)row * DD;
    float sm = 0.f, sv = 0.f;
#pragma unroll 4
    for (int k = lane; k < DD; k += 32) {
        float xv = xr[k];
        sm += xv * muk[k];
        sv += xv * vark[k];
    }
#pragma unroll
    for (int off = 16; off > 0; off >>= 1) {
        sm += __shfl_xor(sm, off, 32);
        sv += __shfl_xor(sv, off, 32);
    }
    if (lane == 0) {
        mu_out[row] = sm + mu_bias[0];
        float v = sv + var_bias[0];
        float sp = (v > 20.f) ? v : log1pf(expf(v));
        d_ws[row] = sp + 1e-8f;
    }
}

// ---------------- cov = (x*rho) @ x^T + cov_bias, diag := d ----------------
// 128x128 C block, Kc=32, 8 waves x (2x4 tiles of 16x16), split-bf16 -> 3 WMMAs/tile/K-step.
// Fill is pure data movement; next tile is prefetched to VGPRs during the WMMA phase using
// running pointers (+BK per step) so the loop carries no 64-bit address rebuilds.
__global__ __launch_bounds__(256) void cov_gemm_kernel(
    const __bf16* __restrict__ xh, const __bf16* __restrict__ xl,
    const __bf16* __restrict__ xrh, const __bf16* __restrict__ xrl,
    const float* __restrict__ cov_bias,
    const float* __restrict__ d_ws,
    float* __restrict__ cov) {
    __shared__ __bf16 As_h[128 * LDA];
    __shared__ __bf16 As_l[128 * LDA];
    __shared__ __bf16 Bs_h[128 * LDA];
    __shared__ __bf16 Bs_l[128 * LDA];

    const int tid    = threadIdx.x;
    const int lane   = tid & 31;
    const int wave   = tid >> 5;
    const int wave_m = wave >> 1;  // 0..3 -> 32-row slab
    const int wave_n = wave & 1;   // 0..1 -> 64-col slab
    const int block_m = blockIdx.y * 128;
    const int block_n = blockIdx.x * 128;

    // per-thread fill coordinates: 2 iterations x 16B (8 bf16) per array;
    // second iteration handled via constant +64 rows = +64*DD elems (64 KiB) offset
    const int fr0 = tid >> 2;          // 0..63
    const int fc  = (tid & 3) * 8;     // 0,8,16,24 (halfs)
    const int o0  = fr0 * LDA + fc;    // LDS offset, iteration 0
    const int o1  = (fr0 + 64) * LDA + fc;

    const __bf16* pah = xrh + (size_t)(block_m + fr0) * DD + fc;
    const __bf16* pal = xrl + (size_t)(block_m + fr0) * DD + fc;
    const __bf16* pbh = xh  + (size_t)(block_n + fr0) * DD + fc;
    const __bf16* pbl = xl  + (size_t)(block_n + fr0) * DD + fc;

    const v8f vzero = {0.f, 0.f, 0.f, 0.f, 0.f, 0.f, 0.f, 0.f};
    v8f acc[2][4];
#pragma unroll
    for (int tm = 0; tm < 2; ++tm)
#pragma unroll
        for (int tn = 0; tn < 4; ++tn) acc[tm][tn] = vzero;

    // prefetch registers for the current K-step's tiles (kk = 0)
    uint4 gah[2], gal[2], gbh[2], gbl[2];
    gah[0] = *(const uint4*)(pah);
    gah[1] = *(const uint4*)(pah + 64 * DD);
    gal[0] = *(const uint4*)(pal);
    gal[1] = *(const uint4*)(pal + 64 * DD);
    gbh[0] = *(const uint4*)(pbh);
    gbh[1] = *(const uint4*)(pbh + 64 * DD);
    gbl[0] = *(const uint4*)(pbl);
    gbl[1] = *(const uint4*)(pbl + 64 * DD);

    for (int kk = 0; kk < DD / BK; ++kk) {
        // ---- commit prefetched tile to LDS ----
        __syncthreads();
        *(uint4*)(As_h + o0) = gah[0];
        *(uint4*)(As_h + o1) = gah[1];
        *(uint4*)(As_l + o0) = gal[0];
        *(uint4*)(As_l + o1) = gal[1];
        *(uint4*)(Bs_h + o0) = gbh[0];
        *(uint4*)(Bs_h + o1) = gbh[1];
        *(uint4*)(Bs_l + o0) = gbl[0];
        *(uint4*)(Bs_l + o1) = gbl[1];
        __syncthreads();

        // ---- issue next tile's global loads (overlap with WMMA below) ----
        if (kk + 1 < DD / BK) {
            pah += BK; pal += BK; pbh += BK; pbl += BK;
            gah[0] = *(const uint4*)(pah);
            gah[1] = *(const uint4*)(pah + 64 * DD);
            gal[0] = *(const uint4*)(pal);
            gal[1] = *(const uint4*)(pal + 64 * DD);
            gbh[0] = *(const uint4*)(pbh);
            gbh[1] = *(const uint4*)(pbh + 64 * DD);
            gbl[0] = *(const uint4*)(pbl);
            gbl[1] = *(const uint4*)(pbl + 64 * DD);
        }

        // ---- WMMA compute ----
        v16bf fa_h[2], fa_l[2];
#pragma unroll
        for (int tm = 0; tm < 2; ++tm) {
            const int row = wave_m * 32 + tm * 16 + (lane & 15);
            fa_h[tm] = load_fragA(As_h + row * LDA, lane);
            fa_l[tm] = load_fragA(As_l + row * LDA, lane);
        }
#pragma unroll
        for (int tn = 0; tn < 4; ++tn) {
            const int col = wave_n * 64 + tn * 16 + (lane & 15);
            v16bf fb_h = load_fragB(Bs_h + col * LDA, lane);
            v16bf fb_l = load_fragB(Bs_l + col * LDA, lane);
#pragma unroll
            for (int tm = 0; tm < 2; ++tm) {
                acc[tm][tn] = __builtin_amdgcn_wmma_f32_16x16x32_bf16(
                    false, fa_h[tm], false, fb_h, (short)0, acc[tm][tn], false, false);
                acc[tm][tn] = __builtin_amdgcn_wmma_f32_16x16x32_bf16(
                    false, fa_h[tm], false, fb_l, (short)0, acc[tm][tn], false, false);
                acc[tm][tn] = __builtin_amdgcn_wmma_f32_16x16x32_bf16(
                    false, fa_l[tm], false, fb_h, (short)0, acc[tm][tn], false, false);
            }
        }
    }

    // ---- epilogue: + cov_bias, diagonal := d[row]; non-temporal stores (output never re-read,
    // keep L2 free for the resident bf16 x tiles) ----
    const float cb = cov_bias[0];
#pragma unroll
    for (int tm = 0; tm < 2; ++tm) {
#pragma unroll
        for (int tn = 0; tn < 4; ++tn) {
            const int row_base = block_m + wave_m * 32 + tm * 16 + ((lane >= 16) ? 8 : 0);
            const int col = block_n + wave_n * 64 + tn * 16 + (lane & 15);
#pragma unroll
            for (int v = 0; v < 8; ++v) {
                const int row = row_base + v;
                float val = acc[tm][tn][v] + cb;
                if (row == col) val = d_ws[row];
                __builtin_nontemporal_store(val, cov + (size_t)row * NN + col);
            }
        }
    }
}

extern "C" void kernel_launch(void* const* d_in, const int* in_sizes, int n_in,
                              void* d_out, int out_size, void* d_ws, size_t ws_size,
                              hipStream_t stream) {
    const float* x     = (const float*)d_in[0];
    const float* mu_k  = (const float*)d_in[1];
    const float* rho_k = (const float*)d_in[2];
    const float* var_k = (const float*)d_in[3];
    const float* mu_b  = (const float*)d_in[4];
    const float* var_b = (const float*)d_in[5];
    const float* cov_b = (const float*)d_in[6];

    float* mu_out = (float*)d_out;        // 8192 floats
    float* cov    = (float*)d_out + NN;   // 8192*8192 floats

    // workspace layout: diag d (32 KB) | xh | xl | xrh | xrl  (4 x 8 MB bf16)
    char* ws = (char*)d_ws;
    float*  dws = (float*)ws;
    __bf16* xh  = (__bf16*)(ws + 32768);
    __bf16* xl  = xh + NXD;
    __bf16* xrh = xl + NXD;
    __bf16* xrl = xrh + NXD;

    prep_kernel<<<NXD / 4 / 256, 256, 0, stream>>>(x, rho_k, xh, xl, xrh, xrl);
    head_kernel<<<NN / 8, 256, 0, stream>>>(x, mu_k, var_k, mu_b, var_b, mu_out, dws);

    dim3 grid(NN / 128, NN / 128);
    cov_gemm_kernel<<<grid, 256, 0, stream>>>(xh, xl, xrh, xrl, cov_b, dws, cov);
}
